// MDRNN_basic_75136157877047
// MI455X (gfx1250) — matrix operand order repaired and checked by reading, not compiled, not run on hardware
//
#include <hip/hip_runtime.h>
#include <hip/hip_bf16.h>

typedef __attribute__((ext_vector_type(16))) _Float16 v16h;
typedef __attribute__((ext_vector_type(8)))  _Float16 v8h;
typedef __attribute__((ext_vector_type(8)))  float    v8f;
typedef __attribute__((ext_vector_type(4)))  unsigned int v4u;
typedef __attribute__((ext_vector_type(8)))  int      v8i_;
typedef __attribute__((ext_vector_type(4)))  int      v4i_;

#define B_TOT 1024
#define C_IN  3
#define H_DIM 32
#define W_DIM 32
#define HID   32
#define G_DIM 160          // 5 * HID
#define NT    10           // G_DIM / 16 N-tiles
#define BT    16           // batch rows per wave (WMMA M)
#define NDIR  4
#define FEAT  128          // 4 * HID
#define NOUT  10

// LDS carve (bytes):
//   Wt_hv [G][32] f16 : 10240
//   Wt_hh [G][32] f16 : 10240
//   Wt_x  [G][32] f16 : 10240 (rows k>=3 zero)
//   h_row [W][16][32] f16 : 32768   (semantic layout, A-fragment friendly)
//   c_row [W][32 lanes][16] f32 : 65536 (opaque per-lane fragment layout)
//   x_stage[2][16][3][32] f32 : 12288 (TDM double buffer)
#define X_STAGE_ELTS (BT * C_IN * W_DIM)            // 1536 floats per buffer
#define LDS_BYTES (3 * G_DIM * HID * 2 + W_DIM * BT * HID * 2 + W_DIM * BT * HID * 4 \
                   + 2 * X_STAGE_ELTS * 4)

__device__ __forceinline__ float fast_sigmoid(float x) {
    return 1.0f / (1.0f + __expf(-x));
}
__device__ __forceinline__ float fast_tanh(float x) {
    float e = __expf(-2.0f * x);
    return (1.0f - e) / (1.0f + e);
}

// Load a 16x32 f16 A-fragment from an LDS block laid out [row16][k32].
// Contiguous K layout (matches B): lanes 0-15 row=lane, K {0..15};
//                                  lanes 16-31 row=lane-16, K {16..31}.
__device__ __forceinline__ v16h load_a_frag(const _Float16* base, int ln, int hi) {
    v8h lo  = *(const v8h*)(base + ln * HID + hi * 16);
    v8h hi8 = *(const v8h*)(base + ln * HID + hi * 16 + 8);
    v16h a;
#pragma unroll
    for (int t = 0; t < 8; ++t) { a[t] = lo[t]; a[8 + t] = hi8[t]; }
    return a;
}

// TDM: async-load one scan row's x slice as a 3D tile into LDS.
//   X: 32 contiguous floats (one image row)
//   Y: 3 channels, stride tensor_dim0_stride = H*W = 1024 elements
//   Z: 16 batch rows, stride tensor_dim1_stride = C*H*W = 3072 elements
// D# packed per CDNA5 ISA (groups 0+1 only).
__device__ __forceinline__ void tdm_load_x_row(unsigned int lds_addr, const float* gptr) {
    unsigned long long ga = (unsigned long long)(uintptr_t)gptr;
    v4u g0;
    g0[0] = 1u;                                   // count=1, is_restore=0, gather=0
    g0[1] = lds_addr;                             // LDS byte address
    g0[2] = (unsigned int)ga;                     // global_addr[31:0]
    g0[3] = ((unsigned int)(ga >> 32) & 0x01FFFFFFu) | 0x80000000u; // addr[56:32] | type=2
    v8i_ g1;
    g1[0] = 0x00020000;                           // workgroup_mask=0, data_size=2 (4B)
    g1[1] = (int)(32u << 16);                     // tensor_dim0[15:0]=32 in [63:48]
    g1[2] = (int)(3u << 16);                      // tensor_dim1[15:0]=3 in [95:80]
    g1[3] = (int)(32u << 16);                     // tile_dim0=32 in [127:112]
    g1[4] = (int)(3u | (16u << 16));              // tile_dim1=3, tile_dim2=16
    g1[5] = 1024;                                 // tensor_dim0_stride[31:0]
    g1[6] = (int)(3072u << 16);                   // tensor_dim1_stride[15:0] in [223:208]
    g1[7] = 0;                                    // tensor_dim1_stride[47:16]
    v4i_ gz = {0, 0, 0, 0};
#if __clang_major__ >= 23
    v8i_ gz8 = {0, 0, 0, 0, 0, 0, 0, 0};
    __builtin_amdgcn_tensor_load_to_lds(g0, g1, gz, gz, gz8, 0);
#else
    __builtin_amdgcn_tensor_load_to_lds(g0, g1, gz, gz, 0);
#endif
}

__global__ __launch_bounds__(32)
void mdlstm2d_scan_kernel(const float* __restrict__ x,     // [B,C,H,W]
                          const float* __restrict__ Wx,    // [NDIR,C,G]
                          const float* __restrict__ Whv,   // [NDIR,HID,G]
                          const float* __restrict__ Whh,   // [NDIR,HID,G]
                          const float* __restrict__ bias,  // [NDIR,G]
                          float* __restrict__ feats)       // [B,FEAT]
{
    extern __shared__ char smem[];
    _Float16* Wt_hv = (_Float16*)smem;               // [G][32], Wt[g][k] = W[k][g]
    _Float16* Wt_hh = Wt_hv + G_DIM * HID;
    _Float16* Wt_x  = Wt_hh + G_DIM * HID;
    _Float16* h_row = Wt_x  + G_DIM * HID;           // [W][BT][HID]
    float*    c_row = (float*)(h_row + W_DIM * BT * HID); // [W][32][16]
    float*    x_stage = c_row + W_DIM * BT * HID;    // [2][BT][C_IN][W] f32

    const int lane = threadIdx.x;       // 0..31 (wave32, one wave per block)
    const int hi   = lane >> 4;         // 0/1
    const int ln   = lane & 15;         // 0..15
    const int tile = blockIdx.x;        // batch tile 0..63
    const int dir  = blockIdx.y;        // 0..3
    const int b0   = tile * BT;
    const int fr   = dir & 1;           // flip rows
    const int fcl  = dir >> 1;          // flip cols

    // Kick off the TDM load of the first scan row's x tile ASAP.
    const float* x_tile_base = x + (size_t)b0 * C_IN * H_DIM * W_DIM;
    {
        int ir0 = fr ? (H_DIM - 1) : 0;
        tdm_load_x_row((unsigned int)(uintptr_t)(x_stage),
                       x_tile_base + (size_t)ir0 * W_DIM);
    }

    // ---- stage weights transposed (f32 -> f16), zero-pad Wx rows k>=C_IN ----
    for (int idx = lane; idx < G_DIM * HID; idx += 32) {
        int g = idx >> 5, k = idx & 31;
        Wt_hv[g * 32 + k] = (_Float16)Whv[(dir * HID + k) * G_DIM + g];
        Wt_hh[g * 32 + k] = (_Float16)Whh[(dir * HID + k) * G_DIM + g];
        Wt_x [g * 32 + k] = (k < C_IN) ? (_Float16)Wx[(dir * C_IN + k) * G_DIM + g]
                                       : (_Float16)0.0f;
    }
    // ---- zero row state (h_up = c_up = 0 for first row) ----
    for (int idx = lane; idx < W_DIM * BT * HID; idx += 32) h_row[idx] = (_Float16)0.0f;
    for (int idx = lane; idx < W_DIM * BT * HID; idx += 32) c_row[idx] = 0.0f;
    // single wave: DS ops are in-order per wave, no barrier needed.

    // bias per N-tile (constant across cells): acc column g = nt*16 + ln
    float bv[NT];
#pragma unroll
    for (int nt = 0; nt < NT; ++nt) bv[nt] = bias[dir * G_DIM + nt * 16 + ln];

    v16h zeroA;
#pragma unroll
    for (int t = 0; t < 16; ++t) zeroA[t] = (_Float16)0.0f;

    const int kbase = hi * 16;   // B-fragment K base for this lane

#pragma unroll 1
    for (int i = 0; i < H_DIM; ++i) {
        // Double-buffered TDM pipeline: issue row i+1, then wait so that the
        // (in-order per wave) row-i tensor load is complete: TENSORcnt <= 1.
        if (i + 1 < H_DIM) {
            int irn = fr ? (H_DIM - 2 - i) : (i + 1);
            tdm_load_x_row((unsigned int)(uintptr_t)(x_stage + ((i + 1) & 1) * X_STAGE_ELTS),
                           x_tile_base + (size_t)irn * W_DIM);
            __builtin_amdgcn_s_wait_tensorcnt(1);
        } else {
            __builtin_amdgcn_s_wait_tensorcnt(0);
        }
        const float* xs = x_stage + (i & 1) * X_STAGE_ELTS;  // [BT][C_IN][W]

        v16h a_left = zeroA;
        float c_left[16];
#pragma unroll
        for (int s = 0; s < 16; ++s) c_left[s] = 0.0f;

#pragma unroll 1
        for (int j = 0; j < W_DIM; ++j) {
            const int jc = fcl ? (W_DIM - 1 - j) : j;

            // h_up / c_up from previous row (still resident in h_row[j]/c_row[j])
            _Float16* hp = h_row + j * BT * HID;
            v16h a_up = load_a_frag(hp, ln, hi);
            float* cp = c_row + (j * 32 + lane) * 16;
            v8f cu0 = *(const v8f*)cp;
            v8f cu1 = *(const v8f*)(cp + 8);

            // A_x: input channels in K=0..2 (lanes 0-15 only), rest zero.
            // Read from the TDM-staged LDS tile (no global traffic in the loop).
            v16h a_x = zeroA;
            if (hi == 0) {
#pragma unroll
                for (int ch = 0; ch < C_IN; ++ch)
                    a_x[ch] = (_Float16)xs[(ln * C_IN + ch) * W_DIM + jc];
            }

            // ---- g = bias + x@Wx + h_up@Whv + h_left@Whh  (10 N-tiles) ----
            v8f acc[NT];
#pragma unroll
            for (int nt = 0; nt < NT; ++nt) {
                v8f a0;
#pragma unroll
                for (int t = 0; t < 8; ++t) a0[t] = bv[nt];
                v16h bhv = *(const v16h*)(Wt_hv + (nt * 16 + ln) * 32 + kbase);
                v16h bhh = *(const v16h*)(Wt_hh + (nt * 16 + ln) * 32 + kbase);
                v16h bwx = *(const v16h*)(Wt_x  + (nt * 16 + ln) * 32 + kbase);
                a0 = __builtin_amdgcn_wmma_f32_16x16x32_f16(false, a_up,   false, bhv, (short)0, a0, false, false);
                a0 = __builtin_amdgcn_wmma_f32_16x16x32_f16(false, a_left, false, bhh, (short)0, a0, false, false);
                a0 = __builtin_amdgcn_wmma_f32_16x16x32_f16(false, a_x,    false, bwx, (short)0, a0, false, false);
                acc[nt] = a0;
            }

            // ---- gates: slot s = p*8+m -> row = m+8*hi, hid = p*16+ln ----
            // gate k occupies N-tiles {2k, 2k+1}; parity p selects hid half.
            float c_new[16], h_new[16];
#pragma unroll
            for (int p = 0; p < 2; ++p) {
#pragma unroll
                for (int m = 0; m < 8; ++m) {
                    const int s = p * 8 + m;
                    float iv = fast_sigmoid(acc[0 + p][m]);
                    float f1 = fast_sigmoid(acc[2 + p][m]);
                    float f2 = fast_sigmoid(acc[4 + p][m]);
                    float gv = fast_tanh   (acc[6 + p][m]);
                    float ov = fast_sigmoid(acc[8 + p][m]);
                    float cu = p ? cu1[m] : cu0[m];
                    float cn = f1 * cu + f2 * c_left[s] + iv * gv;
                    c_new[s] = cn;
                    h_new[s] = ov * fast_tanh(cn);
                }
            }

            // ---- commit state: c in fragment layout (contiguous), h semantic ----
            v8f st0, st1;
#pragma unroll
            for (int m = 0; m < 8; ++m) { st0[m] = c_new[m]; st1[m] = c_new[8 + m]; }
            *(v8f*)cp = st0;
            *(v8f*)(cp + 8) = st1;
#pragma unroll
            for (int p = 0; p < 2; ++p)
#pragma unroll
                for (int m = 0; m < 8; ++m)
                    hp[(m + 8 * hi) * HID + p * 16 + ln] = (_Float16)h_new[p * 8 + m];

            // left-neighbor state for next cell (in-order DS: safe read-after-write)
            a_left = load_a_frag(hp, ln, hi);
#pragma unroll
            for (int s = 0; s < 16; ++s) c_left[s] = c_new[s];
        }
    }

    // Final scan state h(H-1,W-1) == the corner the reference selects for this dir.
#pragma unroll
    for (int t = 0; t < 16; ++t) {
        int hid = hi * 16 + t;
        feats[(size_t)(b0 + ln) * FEAT + dir * HID + hid] =
            (float)h_row[(W_DIM - 1) * BT * HID + ln * HID + hid];
    }
}

__global__ void fc_kernel(const float* __restrict__ feats,
                          const float* __restrict__ fc_w,   // [FEAT,NOUT]
                          const float* __restrict__ fc_b,   // [NOUT]
                          float* __restrict__ out)          // [B,NOUT]
{
    int idx = blockIdx.x * blockDim.x + threadIdx.x;
    if (idx >= B_TOT * NOUT) return;
    int b = idx / NOUT, o = idx % NOUT;
    float s = fc_b[o];
#pragma unroll 4
    for (int k = 0; k < FEAT; ++k) s += feats[b * FEAT + k] * fc_w[k * NOUT + o];
    out[idx] = s;
}

extern "C" void kernel_launch(void* const* d_in, const int* in_sizes, int n_in,
                              void* d_out, int out_size, void* d_ws, size_t ws_size,
                              hipStream_t stream) {
    const float* x    = (const float*)d_in[0];
    const float* Wx   = (const float*)d_in[1];
    const float* Whv  = (const float*)d_in[2];
    const float* Whh  = (const float*)d_in[3];
    const float* bias = (const float*)d_in[4];
    const float* fc_w = (const float*)d_in[5];
    const float* fc_b = (const float*)d_in[6];
    float* out   = (float*)d_out;
    float* feats = (float*)d_ws;   // B_TOT * FEAT floats = 512 KB scratch

    // 138 KB dynamic LDS per (single-wave) workgroup: CDNA5 allows 320 KB/WG.
    (void)hipFuncSetAttribute((const void*)mdlstm2d_scan_kernel,
                              hipFuncAttributeMaxDynamicSharedMemorySize,
                              (int)LDS_BYTES);

    dim3 grid(B_TOT / BT, NDIR);   // 64 batch tiles x 4 directions = 256 waves
    mdlstm2d_scan_kernel<<<grid, 32, LDS_BYTES, stream>>>(x, Wx, Whv, Whh, bias, feats);

    int n = B_TOT * NOUT;
    fc_kernel<<<(n + 255) / 256, 256, 0, stream>>>(feats, fc_w, fc_b, out);
}
